// SloughGPT_85366769975933
// MI455X (gfx1250) — matrix operand order repaired and checked by reading, not compiled, not run on hardware
//
#include <hip/hip_runtime.h>
#include <hip/hip_bf16.h>

// ---------------------------------------------------------------------------
// GPT forward (6 layers) for gfx1250.
//  - all matmuls: v_wmma_f32_16x16x32_bf16
//  - weights pre-transposed+converted to bf16 [N,K] once per launch
//  - GEMM: 128x128 block tile, 8 waves, 32x64/wave (8 WMMA per K-step),
//    double-buffered LDS filled with global_load_async_to_lds_b128,
//    pipelined with s_wait_asynccnt.
// ---------------------------------------------------------------------------

typedef __attribute__((ext_vector_type(16))) __bf16 v16bf;
typedef __attribute__((ext_vector_type(8)))  __bf16 v8bf;
typedef __attribute__((ext_vector_type(8)))  float  v8f;

static constexpr int L_  = 6;
static constexpr int H_  = 1024;
static constexpr int NH_ = 16;
static constexpr int DH_ = 64;
static constexpr int FF_ = 4096;
static constexpr int V_  = 50257;
static constexpr int S_  = 1024;
static constexpr int B_  = 2;
static constexpr int M_  = B_ * S_;   // 2048 token rows

// ---------------------------------------------------------------------------

__device__ __forceinline__ v16bf combine16(v8bf lo, v8bf hi) {
    v16bf r;
#pragma unroll
    for (int i = 0; i < 8; ++i) { r[i] = lo[i]; r[i + 8] = hi[i]; }
    return r;
}

__device__ __forceinline__ v8f wmma_bf16(v16bf a, v16bf b, v8f c) {
    // emits v_wmma_f32_16x16x32_bf16
    return __builtin_amdgcn_wmma_f32_16x16x32_bf16(
        false, a, false, b, (short)0, c, false, false);
}

__device__ __forceinline__ v8f zero_v8f() {
    v8f z;
#pragma unroll
    for (int i = 0; i < 8; ++i) z[i] = 0.0f;
    return z;
}

// async 16-byte copy global -> LDS (per lane), tracked by ASYNCcnt
__device__ __forceinline__ void async_copy_b128(void* lds, const void* gptr) {
    unsigned l = (unsigned)(unsigned long long)lds;
    unsigned long long g = (unsigned long long)gptr;
    asm volatile("global_load_async_to_lds_b128 %0, %1, off"
                 :: "v"(l), "v"(g) : "memory");
}

__device__ __forceinline__ void wait_asynccnt_le2() {
    asm volatile("s_wait_asynccnt 0x2" ::: "memory");
}
__device__ __forceinline__ void wait_asynccnt_0() {
    asm volatile("s_wait_asynccnt 0x0" ::: "memory");
}

// ---------------------------------------------------------------------------
// fp32 -> bf16 conversion (linear layout preserved)
// ---------------------------------------------------------------------------
__global__ __launch_bounds__(256)
void cvt_bf16_kernel(const float* __restrict__ in, __bf16* __restrict__ out,
                     size_t count) {
    size_t i = ((size_t)blockIdx.x * 256 + threadIdx.x) * 4;
#pragma unroll
    for (int j = 0; j < 4; ++j)
        if (i + j < count) out[i + j] = (__bf16)in[i + j];
}

// ---------------------------------------------------------------------------
// fp32 [K,N] -> bf16 [N,K] transpose-convert, per-layer slab via blockIdx.z.
// K, N multiples of 32. Block (32,8); 32x32 tile via LDS.
// ---------------------------------------------------------------------------
__global__ __launch_bounds__(256)
void cvt_transpose_kernel(const float* __restrict__ src, __bf16* __restrict__ dst,
                          int K, int N) {
    __shared__ float tile[32][33];
    const size_t slab = (size_t)blockIdx.z * K * N;
    const float* s = src + slab;
    __bf16*      d = dst + slab;
    const int k0 = blockIdx.x * 32;
    const int n0 = blockIdx.y * 32;
    const int tx = threadIdx.x;
#pragma unroll
    for (int i = threadIdx.y; i < 32; i += 8)
        tile[i][tx] = s[(size_t)(k0 + i) * N + n0 + tx];
    __syncthreads();
#pragma unroll
    for (int i = threadIdx.y; i < 32; i += 8)
        d[(size_t)(n0 + i) * K + k0 + tx] = (__bf16)tile[tx][i];
}

// ---------------------------------------------------------------------------
// Embedding + LayerNorm (block per token; H=1024, 256 threads x 4 elems)
// ---------------------------------------------------------------------------
__global__ __launch_bounds__(256)
void embed_ln_kernel(const int* __restrict__ ids,
                     const float* __restrict__ tok, const float* __restrict__ pos,
                     const float* __restrict__ gam, const float* __restrict__ bet,
                     float* __restrict__ x32, __bf16* __restrict__ xb) {
    const int tokIdx = blockIdx.x;          // [0, M)
    const int s  = tokIdx % S_;
    const int id = ids[tokIdx];
    const int t  = threadIdx.x;
    const int base = t * 4;
    __shared__ float red[16];

    float e[4];
#pragma unroll
    for (int j = 0; j < 4; ++j)
        e[j] = tok[(size_t)id * H_ + base + j] + pos[(size_t)s * H_ + base + j];

    float s1 = 0.f, s2 = 0.f;
#pragma unroll
    for (int j = 0; j < 4; ++j) { s1 += e[j]; s2 += e[j] * e[j]; }
#pragma unroll
    for (int off = 1; off < 32; off <<= 1) {
        s1 += __shfl_xor(s1, off, 32);
        s2 += __shfl_xor(s2, off, 32);
    }
    if ((t & 31) == 0) { red[t >> 5] = s1; red[8 + (t >> 5)] = s2; }
    __syncthreads();
    if (t == 0) {
        float a = 0.f, c = 0.f;
#pragma unroll
        for (int w = 0; w < 8; ++w) { a += red[w]; c += red[8 + w]; }
        float mean = a / (float)H_;
        red[0] = mean;
        red[1] = c / (float)H_ - mean * mean;
    }
    __syncthreads();
    const float mean = red[0];
    const float rstd = rsqrtf(red[1] + 1e-6f);
#pragma unroll
    for (int j = 0; j < 4; ++j) {
        float o = (e[j] - mean) * rstd * gam[base + j] + bet[base + j];
        x32[(size_t)tokIdx * H_ + base + j] = o;
        xb [(size_t)tokIdx * H_ + base + j] = (__bf16)o;
    }
}

// ---------------------------------------------------------------------------
// Residual add + LayerNorm (in-place on the residual stream)
// ---------------------------------------------------------------------------
__global__ __launch_bounds__(256)
void resln_kernel(const float* __restrict__ a, const float* __restrict__ r,
                  const float* __restrict__ gam, const float* __restrict__ bet,
                  float* __restrict__ x32, __bf16* __restrict__ xb) {
    const int tokIdx = blockIdx.x;
    const int t = threadIdx.x;
    const int base = t * 4;
    __shared__ float red[16];

    float e[4];
#pragma unroll
    for (int j = 0; j < 4; ++j) {
        size_t idx = (size_t)tokIdx * H_ + base + j;
        e[j] = a[idx] + r[idx];
    }
    float s1 = 0.f, s2 = 0.f;
#pragma unroll
    for (int j = 0; j < 4; ++j) { s1 += e[j]; s2 += e[j] * e[j]; }
#pragma unroll
    for (int off = 1; off < 32; off <<= 1) {
        s1 += __shfl_xor(s1, off, 32);
        s2 += __shfl_xor(s2, off, 32);
    }
    if ((t & 31) == 0) { red[t >> 5] = s1; red[8 + (t >> 5)] = s2; }
    __syncthreads();
    if (t == 0) {
        float aa = 0.f, cc = 0.f;
#pragma unroll
        for (int w = 0; w < 8; ++w) { aa += red[w]; cc += red[8 + w]; }
        float mean = aa / (float)H_;
        red[0] = mean;
        red[1] = cc / (float)H_ - mean * mean;
    }
    __syncthreads();
    const float mean = red[0];
    const float rstd = rsqrtf(red[1] + 1e-6f);
#pragma unroll
    for (int j = 0; j < 4; ++j) {
        float o = (e[j] - mean) * rstd * gam[base + j] + bet[base + j];
        size_t idx = (size_t)tokIdx * H_ + base + j;
        x32[idx] = o;
        xb [idx] = (__bf16)o;
    }
}

// ---------------------------------------------------------------------------
// Exact GELU, fp32 in -> bf16 out
// ---------------------------------------------------------------------------
__global__ __launch_bounds__(256)
void gelu_kernel(const float* __restrict__ in, __bf16* __restrict__ out,
                 size_t count) {
    size_t i = ((size_t)blockIdx.x * 256 + threadIdx.x) * 4;
#pragma unroll
    for (int j = 0; j < 4; ++j) {
        if (i + j < count) {
            float x = in[i + j];
            float g = 0.5f * x * (1.0f + erff(x * 0.70710678118654752f));
            out[i + j] = (__bf16)g;
        }
    }
}

// ---------------------------------------------------------------------------
// Tiled WMMA GEMM:  out[M,N] = A[M,K](bf16) * W^T + bias,  W stored [N,K] bf16.
// Block: 256 threads (8 waves), tile 128x128, BLK_K = 32, double-buffered LDS
// filled by global_load_async_to_lds_b128 (2 async instrs per wave per tile;
// in-order completion => s_wait_asynccnt 2 guarantees current tile arrived).
// Requires: M,K multiples of 32 and 16B-aligned buffers; rows beyond N read
// adjacent workspace (results masked in epilogue).
// ---------------------------------------------------------------------------
__global__ __launch_bounds__(256)
void gemm_kernel(const __bf16* __restrict__ A, const __bf16* __restrict__ W,
                 const float* __restrict__ bias,
                 float* __restrict__ outF, __bf16* __restrict__ outB,
                 int M, int N, int K) {
    __shared__ __align__(16) __bf16 As[2][128][40];   // [m][k], padded
    __shared__ __align__(16) __bf16 Bs[2][128][40];   // [n][k], padded

    const int t    = threadIdx.x;
    const int lane = t & 31;
    const int wid  = t >> 5;
    const int wm   = wid >> 1;           // 0..3 -> 32-row slice
    const int wn   = wid & 1;            // 0..1 -> 64-col slice
    const int ln   = lane & 15;
    const int g    = lane >> 4;
    const int mblk = blockIdx.y * 128;
    const int nblk = blockIdx.x * 128;

    const int row = t >> 1;              // 0..127
    const int seg = t & 1;               // 0..1 (16 elems each)

    // stage one 128x32 A tile + 128x32 B tile into buffer `buf` (async)
    auto stage = [&](int buf, int kk) {
        async_copy_b128(&As[buf][row][seg * 16],
                        A + (size_t)(mblk + row) * K + kk + seg * 16);
        async_copy_b128(&Bs[buf][row][seg * 16],
                        W + (size_t)(nblk + row) * K + kk + seg * 16);
    };

    v8f acc[2][4];
#pragma unroll
    for (int mf = 0; mf < 2; ++mf)
#pragma unroll
        for (int nf = 0; nf < 4; ++nf) acc[mf][nf] = zero_v8f();

    stage(0, 0);

    int it = 0;
    for (int k0 = 0; k0 < K; k0 += 32, ++it) {
        const int buf = it & 1;
        const bool more = (k0 + 32) < K;
        if (more) {
            stage(buf ^ 1, k0 + 32);     // prefetch next tile (async)
            wait_asynccnt_le2();         // current tile's 2 copies done
        } else {
            wait_asynccnt_0();
        }
        __syncthreads();

        // ---- A fragments (16x32): lane m=ln, K chunks {g*8, g*8+16} ----
        v16bf af[2];
#pragma unroll
        for (int mf = 0; mf < 2; ++mf) {
            int m  = wm * 32 + mf * 16 + ln;
            int ka = g * 8;
            af[mf] = combine16(*(const v8bf*)&As[buf][m][ka],
                               *(const v8bf*)&As[buf][m][ka + 16]);
        }
        // ---- B fragments (32x16): lane n=ln, K chunks {g*8, g*8+16} ----
#pragma unroll
        for (int nf = 0; nf < 4; ++nf) {
            int n  = wn * 64 + nf * 16 + ln;
            int kb = g * 8;
            v16bf bf_ = combine16(*(const v8bf*)&Bs[buf][n][kb],
                                  *(const v8bf*)&Bs[buf][n][kb + 16]);
#pragma unroll
            for (int mf = 0; mf < 2; ++mf)
                acc[mf][nf] = wmma_bf16(af[mf], bf_, acc[mf][nf]);
        }
        __syncthreads();
    }

    // ---- epilogue: C layout row = r + 8*g, col = ln ----
#pragma unroll
    for (int mf = 0; mf < 2; ++mf) {
#pragma unroll
        for (int nf = 0; nf < 4; ++nf) {
#pragma unroll
            for (int r = 0; r < 8; ++r) {
                int gm = mblk + wm * 32 + mf * 16 + r + 8 * g;
                int gn = nblk + wn * 64 + nf * 16 + ln;
                if (gm < M && gn < N) {
                    float v = acc[mf][nf][r] + (bias ? bias[gn] : 0.0f);
                    if (outF) outF[(size_t)gm * N + gn] = v;
                    if (outB) outB[(size_t)gm * N + gn] = (__bf16)v;
                }
            }
        }
    }
}

// ---------------------------------------------------------------------------
// Flash attention: one wave per (batch*head, 16-query tile).
// q/k/v layout: [B, S, NH, DH] bf16. Causal. scale = 1/8.
// ---------------------------------------------------------------------------
__global__ __launch_bounds__(32)
void attn_kernel(const __bf16* __restrict__ qb, const __bf16* __restrict__ kb,
                 const __bf16* __restrict__ vb, __bf16* __restrict__ ob) {
    const int qt   = blockIdx.x;          // query tile [0, S/16)
    const int bh   = blockIdx.y;          // b*NH + h
    const int b    = bh / NH_;
    const int h    = bh % NH_;
    const int lane = threadIdx.x & 31;
    const int ln   = lane & 15;
    const int g    = lane >> 4;

    __shared__ __align__(16) __bf16 Plds[16][40];

    // Q fragments: DH=64 -> two K=32 chunks
    v16bf qf[2];
    {
        const int m = qt * 16 + ln;
        const __bf16* qrow = qb + ((size_t)(b * S_ + m) * NH_ + h) * DH_;
#pragma unroll
        for (int kc = 0; kc < 2; ++kc) {
            int k0 = kc * 32 + g * 8;
            qf[kc] = combine16(*(const v8bf*)(qrow + k0),
                               *(const v8bf*)(qrow + k0 + 16));
        }
    }

    v8f ctx[4];
#pragma unroll
    for (int nt = 0; nt < 4; ++nt) ctx[nt] = zero_v8f();
    float rmax[8], rsum[8];
#pragma unroll
    for (int r = 0; r < 8; ++r) { rmax[r] = -1e30f; rsum[r] = 0.0f; }

    const int nkt = ((qt * 16 + 15) >> 5) + 1;   // 32-key blocks needed

    for (int kt = 0; kt < nkt; ++kt) {
        // ---- scores: Q[16x64] x K^T[64x32] -> two 16x16 tiles ----
        v8f sc[2];
#pragma unroll
        for (int nt = 0; nt < 2; ++nt) {
            v8f c = zero_v8f();
#pragma unroll
            for (int kc = 0; kc < 2; ++kc) {
                int key = kt * 32 + nt * 16 + ln;
                const __bf16* krow =
                    kb + ((size_t)(b * S_ + key) * NH_ + h) * DH_ + kc * 32 + g * 8;
                v16bf kf = combine16(*(const v8bf*)(krow),
                                     *(const v8bf*)(krow + 16));
                c = wmma_bf16(qf[kc], kf, c);
            }
            sc[nt] = c;
        }

        // ---- online softmax (per row r; rows live on 16-lane groups) ----
#pragma unroll
        for (int r = 0; r < 8; ++r) {
            int row = qt * 16 + r + 8 * g;
            int c0  = kt * 32 + ln;
            int c1  = c0 + 16;
            float s0 = sc[0][r] * 0.125f; if (c0 > row) s0 = -1e30f;
            float s1 = sc[1][r] * 0.125f; if (c1 > row) s1 = -1e30f;
            float mloc = fmaxf(s0, s1);
#pragma unroll
            for (int off = 1; off < 16; off <<= 1)
                mloc = fmaxf(mloc, __shfl_xor(mloc, off, 32));
            float mnew  = fmaxf(rmax[r], mloc);
            float scale = __expf(rmax[r] - mnew);
            float p0 = __expf(s0 - mnew);
            float p1 = __expf(s1 - mnew);
            float ps = p0 + p1;
#pragma unroll
            for (int off = 1; off < 16; off <<= 1)
                ps += __shfl_xor(ps, off, 32);
            rsum[r] = rsum[r] * scale + ps;
            rmax[r] = mnew;
#pragma unroll
            for (int nt = 0; nt < 4; ++nt) ctx[nt][r] *= scale;
            Plds[r + 8 * g][ln]      = (__bf16)p0;
            Plds[r + 8 * g][ln + 16] = (__bf16)p1;
        }
        __syncthreads();

        // ---- P fragment (16x32) from LDS ----
        v16bf pf;
        {
            int k0 = g * 16;
            pf = combine16(*(const v8bf*)&Plds[ln][k0],
                           *(const v8bf*)&Plds[ln][k0 + 8]);
        }
        // ---- P[16x32] x V[32x64] into 4 ctx tiles ----
#pragma unroll
        for (int nt = 0; nt < 4; ++nt) {
            v16bf vf;
            int dh  = nt * 16 + ln;
            int kbs = kt * 32 + g * 16;
#pragma unroll
            for (int j = 0; j < 16; ++j)
                vf[j] = vb[((size_t)(b * S_ + kbs + j) * NH_ + h) * DH_ + dh];
            ctx[nt] = wmma_bf16(pf, vf, ctx[nt]);
        }
        __syncthreads();
    }

    // ---- normalize + store bf16 context [B,S,NH,DH] ----
#pragma unroll
    for (int nt = 0; nt < 4; ++nt) {
#pragma unroll
        for (int r = 0; r < 8; ++r) {
            int row = qt * 16 + r + 8 * g;
            int dh  = nt * 16 + ln;
            float v = ctx[nt][r] / rsum[r];
            ob[((size_t)(b * S_ + row) * NH_ + h) * DH_ + dh] = (__bf16)v;
        }
    }
}

// ---------------------------------------------------------------------------
// Host orchestration
// ---------------------------------------------------------------------------
extern "C" void kernel_launch(void* const* d_in, const int* in_sizes, int n_in,
                              void* d_out, int out_size, void* d_ws, size_t ws_size,
                              hipStream_t stream) {
    (void)in_sizes; (void)n_in; (void)out_size; (void)ws_size;

    const int*   ids  = (const int*)  d_in[0];
    const float* tok  = (const float*)d_in[1];
    const float* pos  = (const float*)d_in[2];
    const float* ln0g = (const float*)d_in[3];
    const float* ln0b = (const float*)d_in[4];
    const float* Wq   = (const float*)d_in[5];
    const float* bq   = (const float*)d_in[6];
    const float* Wk   = (const float*)d_in[7];
    const float* bk   = (const float*)d_in[8];
    const float* Wv   = (const float*)d_in[9];
    const float* bv   = (const float*)d_in[10];
    const float* Wo   = (const float*)d_in[11];
    const float* bo   = (const float*)d_in[12];
    const float* ln1g = (const float*)d_in[13];
    const float* ln1b = (const float*)d_in[14];
    const float* Wi   = (const float*)d_in[15];
    const float* bi   = (const float*)d_in[16];
    const float* Wf   = (const float*)d_in[17];
    const float* bfp  = (const float*)d_in[18];
    const float* ln2g = (const float*)d_in[19];
    const float* ln2b = (const float*)d_in[20];

    char*  ws  = (char*)d_ws;
    size_t off = 0;
    auto carve = [&](size_t bytes) -> char* {
        char* p = ws + off;
        off += (bytes + 255) & ~(size_t)255;
        return p;
    };

    // all weight copies stored [N,K] bf16 (transposed where needed)
    __bf16* tokB = (__bf16*)carve((size_t)V_ * H_ * 2);   // [V,H] already N,K
    __bf16* WqB  = (__bf16*)carve((size_t)L_ * H_ * H_ * 2);
    __bf16* WkB  = (__bf16*)carve((size_t)L_ * H_ * H_ * 2);
    __bf16* WvB  = (__bf16*)carve((size_t)L_ * H_ * H_ * 2);
    __bf16* WoB  = (__bf16*)carve((size_t)L_ * H_ * H_ * 2);
    __bf16* WiB  = (__bf16*)carve((size_t)L_ * H_ * FF_ * 2);  // [FF,H]
    __bf16* WfB  = (__bf16*)carve((size_t)L_ * FF_ * H_ * 2);  // [H,FF]
    float*  x32  = (float*) carve((size_t)M_ * H_ * 4);
    __bf16* xb   = (__bf16*)carve((size_t)M_ * H_ * 2);
    __bf16* qbuf = (__bf16*)carve((size_t)M_ * H_ * 2);
    __bf16* kbuf = (__bf16*)carve((size_t)M_ * H_ * 2);
    __bf16* vbuf = (__bf16*)carve((size_t)M_ * H_ * 2);
    __bf16* ctxb = (__bf16*)carve((size_t)M_ * H_ * 2);
    float*  att32= (float*) carve((size_t)M_ * H_ * 4);
    float*  ff1  = (float*) carve((size_t)M_ * FF_ * 4);
    __bf16* hb   = (__bf16*)carve((size_t)M_ * FF_ * 2);
    float*  ffn32= (float*) carve((size_t)M_ * H_ * 4);
    (void)carve(4096);  // guard region: OOB async reads in last LM-head block

    // --- weight pack: convert (+transpose [K,N] -> [N,K]) once per launch ---
    cvt_bf16_kernel<<<dim3((unsigned)(((size_t)V_ * H_ + 1023) / 1024)), 256, 0,
                      stream>>>(tok, tokB, (size_t)V_ * H_);
    const dim3 tHH(H_ / 32, H_ / 32, L_);
    const dim3 tHF(H_ / 32, FF_ / 32, L_);   // src K=H, N=FF
    const dim3 tFH(FF_ / 32, H_ / 32, L_);   // src K=FF, N=H
    cvt_transpose_kernel<<<tHH, dim3(32, 8), 0, stream>>>(Wq, WqB, H_, H_);
    cvt_transpose_kernel<<<tHH, dim3(32, 8), 0, stream>>>(Wk, WkB, H_, H_);
    cvt_transpose_kernel<<<tHH, dim3(32, 8), 0, stream>>>(Wv, WvB, H_, H_);
    cvt_transpose_kernel<<<tHH, dim3(32, 8), 0, stream>>>(Wo, WoB, H_, H_);
    cvt_transpose_kernel<<<tHF, dim3(32, 8), 0, stream>>>(Wi, WiB, H_, FF_);
    cvt_transpose_kernel<<<tFH, dim3(32, 8), 0, stream>>>(Wf, WfB, FF_, H_);

    // embeddings + ln0
    embed_ln_kernel<<<dim3(M_), 256, 0, stream>>>(ids, tok, pos, ln0g, ln0b, x32, xb);

    const dim3 gHH((H_ + 127) / 128, (M_ + 127) / 128);
    const dim3 gHF((FF_ + 127) / 128, (M_ + 127) / 128);
    const dim3 gAT(S_ / 16, B_ * NH_);

    for (int l = 0; l < L_; ++l) {
        const __bf16* wq = WqB + (size_t)l * H_ * H_;
        const __bf16* wk = WkB + (size_t)l * H_ * H_;
        const __bf16* wv = WvB + (size_t)l * H_ * H_;
        const __bf16* wo = WoB + (size_t)l * H_ * H_;
        const __bf16* wi = WiB + (size_t)l * H_ * FF_;
        const __bf16* wf = WfB + (size_t)l * FF_ * H_;

        gemm_kernel<<<gHH, 256, 0, stream>>>(xb, wq, bq + (size_t)l * H_,
                                             nullptr, qbuf, M_, H_, H_);
        gemm_kernel<<<gHH, 256, 0, stream>>>(xb, wk, bk + (size_t)l * H_,
                                             nullptr, kbuf, M_, H_, H_);
        gemm_kernel<<<gHH, 256, 0, stream>>>(xb, wv, bv + (size_t)l * H_,
                                             nullptr, vbuf, M_, H_, H_);

        attn_kernel<<<gAT, 32, 0, stream>>>(qbuf, kbuf, vbuf, ctxb);

        gemm_kernel<<<gHH, 256, 0, stream>>>(ctxb, wo, bo + (size_t)l * H_,
                                             att32, nullptr, M_, H_, H_);
        resln_kernel<<<dim3(M_), 256, 0, stream>>>(att32, x32,
                                                   ln1g + (size_t)l * H_,
                                                   ln1b + (size_t)l * H_, x32, xb);

        gemm_kernel<<<gHF, 256, 0, stream>>>(xb, wi, bi + (size_t)l * FF_,
                                             ff1, nullptr, M_, FF_, H_);
        gelu_kernel<<<dim3((unsigned)(((size_t)M_ * FF_ + 1023) / 1024)), 256, 0, stream>>>(
            ff1, hb, (size_t)M_ * FF_);
        gemm_kernel<<<gHH, 256, 0, stream>>>(hb, wf, bfp + (size_t)l * H_,
                                             ffn32, nullptr, M_, H_, FF_);
        resln_kernel<<<dim3(M_), 256, 0, stream>>>(ffn32, x32,
                                                   ln2g + (size_t)l * H_,
                                                   ln2b + (size_t)l * H_, x32, xb);
    }

    // tied LM head: logits = x @ tok_emb^T   (tok_emb already [N=V, K=H])
    const dim3 gLM((V_ + 127) / 128, (M_ + 127) / 128);
    gemm_kernel<<<gLM, 256, 0, stream>>>(xb, tokB, nullptr,
                                         (float*)d_out, nullptr, M_, V_, H_);
}
